// MACUnit_51582557225362
// MI455X (gfx1250) — matrix operand order repaired and checked by reading, not compiled, not run on hardware
//
#include <hip/hip_runtime.h>
#include <stdint.h>

typedef __bf16 bf16;
typedef __bf16 v16bf __attribute__((ext_vector_type(16)));
typedef __bf16 v8bf  __attribute__((ext_vector_type(8)));
typedef float  v8f   __attribute__((ext_vector_type(8)));
typedef unsigned int v4u __attribute__((ext_vector_type(4)));
typedef int v4i __attribute__((ext_vector_type(4)));
typedef int v8i __attribute__((ext_vector_type(8)));

#ifndef __has_builtin
#define __has_builtin(x) 0
#endif
#if __has_builtin(__builtin_amdgcn_tensor_load_to_lds)
#define HAVE_TDM 1
#else
#define HAVE_TDM 0
#endif

#define DEV __device__ __forceinline__

// ---------------- CDNA5 helpers ----------------
DEV void async_g2l_b128(void* ldsp, const void* gptr) {
  // LDS aperture: addr[31:0] is the LDS offset, so truncation is correct.
  uint32_t l = (uint32_t)(uintptr_t)ldsp;
  asm volatile("global_load_async_to_lds_b128 %0, %1, off"
               :: "v"(l), "v"(gptr) : "memory");
}
DEV void wait_async0() { asm volatile("s_wait_asynccnt 0" ::: "memory"); }
DEV void wait_async8() { asm volatile("s_wait_asynccnt 8" ::: "memory"); }

DEV float wave_sum(float x){ for (int o=16;o;o>>=1) x += __shfl_xor(x,o,32); return x; }
DEV float wave_max(float x){ for (int o=16;o;o>>=1) x = fmaxf(x,__shfl_xor(x,o,32)); return x; }

// TDM: 1-row 2D tile of `words` dwords, global -> LDS (issue from ONE wave; EXEC ignored).
DEV void tdm_load_row_to_lds(void* ldsp, const void* gptr, int words) {
#if HAVE_TDM
  const uint64_t ga = (uint64_t)(uintptr_t)gptr;
  const uint32_t la = (uint32_t)(uintptr_t)ldsp;
  v4u g0 = {0u, 0u, 0u, 0u};
  g0[0] = 1u;                              // count=1, user descriptor
  g0[1] = la;                              // lds_addr
  g0[2] = (uint32_t)ga;                    // global_addr[31:0]
  g0[3] = (uint32_t)((ga >> 32) & 0x01FFFFFFu) | (2u << 30);  // addr[56:32] | type=2
  v8i g1 = {0,0,0,0,0,0,0,0};
  g1[0] = (2 << 16);                       // data_size=4B; no multicast/pad/iterate
  g1[1] = (words & 0xFFFF) << 16;          // tensor_dim0[15:0]
  g1[2] = ((words >> 16) & 0xFFFF) | (1 << 16);   // tensor_dim0 hi | tensor_dim1=1
  g1[3] = (words & 0xFFFF) << 16;          // tile_dim0 = words
  g1[4] = 1;                               // tile_dim1 = 1
  g1[5] = words;                           // tensor_dim0_stride lo
  g1[6] = (words & 0xFFFF) << 16;          // tensor_dim1_stride lo16
  v4i gz = {0,0,0,0};
#if defined(__clang_major__) && (__clang_major__ >= 23)
  v8i gz8 = {0,0,0,0,0,0,0,0};
  __builtin_amdgcn_tensor_load_to_lds(g0, g1, gz, gz, gz8, 0);
#else
  __builtin_amdgcn_tensor_load_to_lds(g0, g1, gz, gz, 0);
#endif
  __builtin_amdgcn_s_wait_tensorcnt((short)0);
#else
  (void)ldsp; (void)gptr; (void)words;
#endif
}

// ---------------- generic bf16 WMMA GEMM ----------------
// C[m,n] = alpha * ( sum_k A[m,k]*B[n,k] + bias[n] )   (A @ B^T, B stored [N,K])
// 128 threads = 4 waves; 64x64 tile; wave w computes rows [16w,16w+16) x all 64 cols.
// REQUIRES: full 64x64 tiles and K%64==0 -- true at every call site (K in {64,512,1024}).
// 64-deep K slabs staged with double-buffered async global->LDS copies (ASYNCcnt pipeline);
// per slab each wave runs 2 x (10 ds_load_b128 -> 4 back-to-back v_wmma).
__global__ void wmma_gemm_kernel(
    const bf16* __restrict__ A, int lda,
    const bf16* __restrict__ B, int ldb,
    const float* __restrict__ bias,
    float* __restrict__ Cf, bf16* __restrict__ Cbf, int ldc,
    int M, int N, int K, float alpha,
    long saz, long sbz, long scz)
{
  __shared__ bf16 At[2][64][72];   // 144B row stride: 16B-aligned, conflict-free (36r mod 64)
  __shared__ bf16 Bt[2][64][72];
  const int z = blockIdx.z;
  A += saz * z; B += sbz * z;
  const long coff = scz * z;
  const int tileN = blockIdx.x * 64, tileM = blockIdx.y * 64;
  const int tid  = threadIdx.x;
  const int lane = tid & 31, wv = tid >> 5;
  const int hf = lane >> 4, l15 = lane & 15;
  const int ldr = tid >> 1, ldcol = (tid & 1) * 32;

  const bf16* Arow = A + (long)(tileM + ldr) * lda + ldcol;
  const bf16* Brow = B + (long)(tileN + ldr) * ldb + ldcol;

  v8f acc[4];
  v8f zero = {};
  #pragma unroll
  for (int j = 0; j < 4; ++j) acc[j] = zero;

  auto stage = [&](int buf, int k0) {   // 8 async b128 per thread: 32 cols of A + 32 of B
    #pragma unroll
    for (int c = 0; c < 4; ++c) {
      async_g2l_b128(&At[buf][ldr][ldcol + c * 8], Arow + k0 + c * 8);
      async_g2l_b128(&Bt[buf][ldr][ldcol + c * 8], Brow + k0 + c * 8);
    }
  };
  auto compute = [&](int cb) {
    #pragma unroll
    for (int kh = 0; kh < 2; ++kh) {    // two K=32 halves of the 64-deep slab
      // A fragment: ISA 16-bit A 16x32 layout. lane m=l15;
      // half 0: e0-7 -> K0-7, e8-15 -> K16-23 ; half 1: e0-7 -> K8-15, e8-15 -> K24-31
      v16bf afrag;
      const v8bf* arow = (const v8bf*)&At[cb][wv * 16 + l15][kh * 32];
      ((v8bf*)&afrag)[0] = arow[hf];
      ((v8bf*)&afrag)[1] = arow[2 + hf];
      // B fragments first, then 4 WMMAs back-to-back (distinct accumulators: no hazard)
      v16bf bfr[4];
      #pragma unroll
      for (int j = 0; j < 4; ++j) {
        const v8bf* brow = (const v8bf*)&Bt[cb][j * 16 + l15][kh * 32];
        ((v8bf*)&bfr[j])[0] = brow[hf];
        ((v8bf*)&bfr[j])[1] = brow[2 + hf];
      }
      #pragma unroll
      for (int j = 0; j < 4; ++j)
        acc[j] = __builtin_amdgcn_wmma_f32_16x16x32_bf16(
            false, afrag, false, bfr[j], (short)0, acc[j], false, false);
    }
  };

  const int nt = K >> 6;   // K/64 slabs
  stage(0, 0);
  for (int kt = 0; kt < nt - 1; ++kt) {          // steady state: always prefetching
    stage((kt + 1) & 1, (kt + 1) << 6);
    wait_async8();                               // retire slab kt, keep 8 in flight
    __syncthreads();
    compute(kt & 1);
    __syncthreads();                             // protect buffer before next overwrite
  }
  wait_async0();                                 // tail slab
  __syncthreads();
  compute((nt - 1) & 1);

  // epilogue: C/D f32 layout -> element (m = 16w + 8*hf + r, n = 16j + l15)
  #pragma unroll
  for (int j = 0; j < 4; ++j) {
    const int n = tileN + j * 16 + l15;
    const float bv = (bias && n < N) ? bias[n] : 0.f;
    #pragma unroll
    for (int r = 0; r < 8; ++r) {
      const int m = tileM + wv * 16 + hf * 8 + r;
      if (m < M && n < N) {
        float v = alpha * (acc[j][r] + bv);
        if (Cf)  Cf [coff + (long)m * ldc + n] = v;
        if (Cbf) Cbf[coff + (long)m * ldc + n] = (bf16)v;
      }
    }
  }
}

// ---------------- conversion kernels ----------------
__global__ void cast_kernel(const float* __restrict__ s, bf16* __restrict__ d, long n) {
  for (long i = (long)blockIdx.x * blockDim.x + threadIdx.x; i < n;
       i += (long)gridDim.x * blockDim.x)
    d[i] = (bf16)s[i];
}

// dst[c, r] = src[r, c] (per-z slab of R*C elements)
__global__ void tcast_kernel(const float* __restrict__ src, bf16* __restrict__ dst,
                             int R, int C) {
  __shared__ float tile[32][33];
  const long zo = (long)blockIdx.z * R * C;
  const int c0 = blockIdx.x * 32, r0 = blockIdx.y * 32;
  const int tx = threadIdx.x & 31, ty = threadIdx.x >> 5;   // 256 thr: ty 0..7
  for (int i = ty; i < 32; i += 8) {
    int r = r0 + i, c = c0 + tx;
    tile[i][tx] = (r < R && c < C) ? src[zo + (long)r * C + c] : 0.f;
  }
  __syncthreads();
  for (int i = ty; i < 32; i += 8) {
    int c = c0 + i, r = r0 + tx;
    if (c < C && r < R) dst[zo + (long)c * R + r] = (bf16)tile[tx][i];
  }
}

// ---------------- per-step small kernels ----------------
__global__ void init_state_kernel(const float* __restrict__ c0, const float* __restrict__ m0,
                                  float* __restrict__ control, float* __restrict__ memory,
                                  bf16* __restrict__ cqA, bf16* __restrict__ wcA) {
  int i = blockIdx.x * blockDim.x + threadIdx.x;
  if (i < 64 * 512) {
    int d = i & 511, b = i >> 9;
    float cv = c0[d], mv = m0[d];
    control[i] = cv; memory[i] = mv;
    cqA[(long)b * 1024 + d] = (bf16)cv;
    wcA[(long)b * 1024 + 512 + d] = (bf16)mv;
  }
}

// aw = softmax_s( (cq*ctx)·attn_w + b ); control = sum_s aw*ctx   (one block per b)
__global__ void control_attn_kernel(const float* __restrict__ cq, const float* __restrict__ ctx,
                                    const float* __restrict__ aww, const float* __restrict__ awb,
                                    float* __restrict__ control, bf16* __restrict__ control_bf,
                                    bf16* __restrict__ cqA) {
  const int b = blockIdx.x, tid = threadIdx.x, lane = tid & 31, wv = tid >> 5;
  __shared__ float p[32];
  const float* cqb = cq + (long)b * 512;
  const float* cb  = ctx + (long)b * 32 * 512;
  for (int s = wv; s < 32; s += 8) {
    float acc = 0.f;
    #pragma unroll
    for (int j = 0; j < 16; ++j) {
      int d = lane * 16 + j;
      acc += cqb[d] * cb[(long)s * 512 + d] * aww[d];
    }
    acc = wave_sum(acc);
    if (lane == 0) p[s] = acc + awb[0];
  }
  __syncthreads();
  if (tid < 32) {
    float x = p[tid];
    float m = wave_max(x);
    float e = __expf(x - m);
    float s = wave_sum(e);
    p[tid] = e / s;
  }
  __syncthreads();
  for (int d = tid; d < 512; d += 256) {
    float acc = 0.f;
    for (int s = 0; s < 32; ++s) acc += p[s] * cb[(long)s * 512 + d];
    control[(long)b * 512 + d] = acc;
    control_bf[(long)b * 512 + d] = (bf16)acc;
    cqA[(long)b * 1024 + d] = (bf16)acc;   // feeds next step's cq concat
  }
}

// scores [64,8,512] -> softmax over l per head, mean over heads -> mask [64,512]
__global__ void mask_softmax_kernel(const float* __restrict__ scores, float* __restrict__ mask) {
  const int b = blockIdx.x, tid = threadIdx.x, lane = tid & 31, wv = tid >> 5;
  __shared__ float red[8][512];
  const float* sc = scores + ((long)b * 8 + wv) * 512;
  float v[16]; float m = -3.4e38f;
  #pragma unroll
  for (int j = 0; j < 16; ++j) { v[j] = sc[lane * 16 + j]; m = fmaxf(m, v[j]); }
  m = wave_max(m);
  float ssum = 0.f;
  #pragma unroll
  for (int j = 0; j < 16; ++j) { v[j] = __expf(v[j] - m); ssum += v[j]; }
  ssum = wave_sum(ssum);
  const float inv = 0.125f / ssum;          // fold mean over 8 heads
  #pragma unroll
  for (int j = 0; j < 16; ++j) red[wv][lane * 16 + j] = v[j] * inv;
  __syncthreads();
  for (int l = tid; l < 512; l += 256) {
    float s = 0.f;
    #pragma unroll
    for (int w = 0; w < 8; ++w) s += red[w][l];
    mask[(long)b * 512 + l] = s;
  }
}

__global__ void mk_v_kernel(const float* __restrict__ control, const float* __restrict__ rc_mask,
                            const float* __restrict__ aww, bf16* __restrict__ v) {
  int i = blockIdx.x * blockDim.x + threadIdx.x;
  if (i < 64 * 512) { int d = i & 511; v[i] = (bf16)(control[i] * rc_mask[i] * aww[d]); }
}
__global__ void mk_t_kernel(const float* __restrict__ know_mask, const float* __restrict__ mem_raw,
                            const float* __restrict__ mem_mask, const float* __restrict__ u,
                            bf16* __restrict__ t) {
  int i = blockIdx.x * blockDim.x + threadIdx.x;
  if (i < 64 * 512) {
    int b = i >> 9, d = i & 511;
    t[i] = (bf16)(know_mask[i] * mem_raw[i] * mem_mask[i] * u[(long)b * 1024 + d]);
  }
}
__global__ void add_g_kernel(const float* __restrict__ g0, const float* __restrict__ u,
                             float* __restrict__ g) {
  int i = blockIdx.x * blockDim.x + threadIdx.x;
  if (i < 64 * 512) { int b = i >> 9, d = i & 511; g[i] = g0[i] + u[(long)b * 1024 + 512 + d]; }
}
__global__ void mem_update_kernel(const float* __restrict__ wc, const float* __restrict__ wc_mask,
                                  float* __restrict__ memory, bf16* __restrict__ wcA) {
  int i = blockIdx.x * blockDim.x + threadIdx.x;
  if (i < 64 * 512) {
    int b = i >> 9, d = i & 511;
    float mv = wc[i] * wc_mask[i];
    memory[i] = mv;
    wcA[(long)b * 1024 + 512 + d] = (bf16)mv;   // memory_prev for next step's write concat
  }
}

// ---------------- knowledge streaming: logits -> softmax -> weighted read ----------------
// One block (8 waves) per batch b. know_t [64,1024,512] bf16 is L2-resident (67MB < 192MB).
__global__ void read_unit_kernel(const bf16* __restrict__ kt, const float* __restrict__ g,
                                 bf16* __restrict__ wcA) {
  const int b = blockIdx.x, tid = threadIdx.x, lane = tid & 31, wv = tid >> 5;
  __shared__ float lg[1024];
  __shared__ float red[8][512];
  __shared__ float gld[512];
  __shared__ float sred[8];

  // stage g[b] (2KB) into LDS: Tensor Data Mover if available, else async b128 copies
#if HAVE_TDM
  if (wv == 0) tdm_load_row_to_lds(&gld[0], g + (long)b * 512, 512);
#else
  if (tid < 128) async_g2l_b128(&gld[tid * 4], g + (long)b * 512 + tid * 4);
  wait_async0();
#endif
  __syncthreads();

  float gv[16];
  #pragma unroll
  for (int j = 0; j < 16; ++j) gv[j] = gld[lane * 16 + j];
  const bf16* base = kt + (long)b * 1024 * 512;

  // phase 1: logits[k] = know_t[b,k,:] . g[b]   (wave-per-row, lane owns 16 contiguous d)
  for (int k = wv * 128; k < wv * 128 + 128; ++k) {
    const v8bf* row = (const v8bf*)(base + (long)k * 512 + lane * 16);
    if (k + 1 < 1024) __builtin_prefetch(base + (long)(k + 1) * 512 + lane * 16, 0, 3);
    v8bf x0 = row[0], x1 = row[1];
    float s = 0.f;
    #pragma unroll
    for (int j = 0; j < 8; ++j) s += (float)x0[j] * gv[j];
    #pragma unroll
    for (int j = 0; j < 8; ++j) s += (float)x1[j] * gv[8 + j];
    s = wave_sum(s);
    if (lane == 0) lg[k] = s;
  }
  __syncthreads();

  // phase 2: softmax over 1024 (unnormalized exp in lg, broadcast 1/sum)
  float m = -3.4e38f;
  for (int i = tid; i < 1024; i += 256) m = fmaxf(m, lg[i]);
  m = wave_max(m);
  if (lane == 0) sred[wv] = m;
  __syncthreads();
  m = sred[0];
  #pragma unroll
  for (int w = 1; w < 8; ++w) m = fmaxf(m, sred[w]);
  float ssum = 0.f;
  for (int i = tid; i < 1024; i += 256) { float e = __expf(lg[i] - m); lg[i] = e; ssum += e; }
  ssum = wave_sum(ssum);
  __syncthreads();
  if (lane == 0) sred[wv] = ssum;
  __syncthreads();
  float tot = 0.f;
  #pragma unroll
  for (int w = 0; w < 8; ++w) tot += sred[w];
  const float inv = 1.f / tot;

  // phase 3: read = sum_k p[k] * know_t[b,k,:]
  float acc[16];
  #pragma unroll
  for (int j = 0; j < 16; ++j) acc[j] = 0.f;
  for (int k = wv * 128; k < wv * 128 + 128; ++k) {
    const float p = lg[k];
    const v8bf* row = (const v8bf*)(base + (long)k * 512 + lane * 16);
    v8bf x0 = row[0], x1 = row[1];
    #pragma unroll
    for (int j = 0; j < 8; ++j) acc[j] += p * (float)x0[j];
    #pragma unroll
    for (int j = 0; j < 8; ++j) acc[8 + j] += p * (float)x1[j];
  }
  #pragma unroll
  for (int j = 0; j < 16; ++j) red[wv][lane * 16 + j] = acc[j];
  __syncthreads();
  for (int d = tid; d < 512; d += 256) {
    float s = 0.f;
    #pragma unroll
    for (int w = 0; w < 8; ++w) s += red[w][d];
    wcA[(long)b * 1024 + d] = (bf16)(s * inv);   // read_bf for write-unit concat
  }
}

// ---------------- host orchestration ----------------
// Flattened params: top-level dict order {context,question,knowledge,params};
// params pytree-flattened with sorted dict keys at each level.
enum {
  IN_CONTEXT = 0, IN_QUESTION, IN_KNOWLEDGE,
  P_CONTROL0,
  P_C_ATTN_B, P_C_ATTN_W, P_C_CQ_B, P_C_CQ_W, P_C_POS_B, P_C_POS_W,
  P_MEM0,
  P_R_ATTN_B, P_R_ATTN_W, P_R_CONCAT_B,
  P_R_CM_BK, P_R_CM_BQ, P_R_CM_WK, P_R_CM_WQ,
  P_R_CONCAT_W, P_R_CPM_B, P_R_CPM_W, P_R_KNOW_B,
  P_R_KM_BK, P_R_KM_BQ, P_R_KM_WK, P_R_KM_WQ,
  P_R_KNOW_W, P_R_MEM_B,
  P_R_MM_BK, P_R_MM_BQ, P_R_MM_WK, P_R_MM_WQ,
  P_R_MEM_W,
  P_W_CONCAT_B,
  P_W_CM_BK, P_W_CM_BQ, P_W_CM_WK, P_W_CM_WQ,
  P_W_CONCAT_W, P_W_CPM_B, P_W_CPM_W,
  N_INPUTS
};

static void launch_gemm(hipStream_t st, const bf16* A, int lda, const bf16* B, int ldb,
                        const float* bias, float* Cf, bf16* Cbf, int ldc,
                        int M, int N, int K, float alpha,
                        int Z = 1, long saz = 0, long sbz = 0, long scz = 0) {
  dim3 grid((N + 63) / 64, (M + 63) / 64, Z), blk(128);
  wmma_gemm_kernel<<<grid, blk, 0, st>>>(A, lda, B, ldb, bias, Cf, Cbf, ldc,
                                         M, N, K, alpha, saz, sbz, scz);
}

extern "C" void kernel_launch(void* const* d_in, const int* in_sizes, int n_in,
                              void* d_out, int out_size, void* d_ws, size_t ws_size,
                              hipStream_t stream) {
  (void)in_sizes; (void)n_in; (void)out_size; (void)ws_size;
  const float* F[N_INPUTS];
  for (int i = 0; i < N_INPUTS; ++i) F[i] = (const float*)d_in[i];

  // ---- workspace allocator ----
  char* wsp = (char*)d_ws; size_t cur = 0;
  auto alloc = [&](size_t bytes) -> void* {
    void* p = wsp + cur; cur = (cur + bytes + 255) & ~(size_t)255; return p;
  };
  const long BD = 64 * 512;  // batch x dim elements

  bf16* kt       = (bf16*)alloc(64L * 1024 * 512 * 2);   // knowledge^T bf16 (L2-resident)
  bf16* q_bf     = (bf16*)alloc(64L * 1024 * 2);
  bf16* posw_bf  = (bf16*)alloc(12L * 512 * 1024 * 2);
  bf16* cqw_bf   = (bf16*)alloc(512L * 1024 * 2);
  bf16* memw_bf  = (bf16*)alloc(512L * 512 * 2);
  bf16* knoww_bf = (bf16*)alloc(512L * 512 * 2);
  bf16* rcw_bf   = (bf16*)alloc(512L * 1024 * 2);
  bf16* rcwT_bf  = (bf16*)alloc(1024L * 512 * 2);
  bf16* kwT_bf   = (bf16*)alloc(512L * 512 * 2);
  bf16* rcpm_bf  = (bf16*)alloc(512L * 1024 * 2);
  bf16* wcw_bf   = (bf16*)alloc(512L * 1024 * 2);
  bf16* wcpmw_bf = (bf16*)alloc(512L * 1024 * 2);
  bf16* wq_bf[4], *wk_bf[4], *kX_bf[4];
  for (int i = 0; i < 4; ++i) wq_bf[i] = (bf16*)alloc(512L * 512 * 2);
  for (int i = 0; i < 4; ++i) wk_bf[i] = (bf16*)alloc(512L * 512 * 2);
  bf16* cpm_bf  = (bf16*)alloc(512L * 512 * 2);
  bf16* wcpm_bf = (bf16*)alloc(512L * 512 * 2);
  for (int i = 0; i < 4; ++i) kX_bf[i] = (bf16*)alloc(512L * 512 * 2);
  float* control   = (float*)alloc(BD * 4);
  bf16*  control_bf= (bf16*) alloc(BD * 2);
  float* cq_f      = (float*)alloc(BD * 4);
  bf16*  cqA       = (bf16*) alloc(64L * 1024 * 2);   // [control_bf | pa_bf]
  float* memory_f  = (float*)alloc(BD * 4);
  bf16*  wcA       = (bf16*) alloc(64L * 1024 * 2);   // [read_bf | memory_bf]
  bf16*  qproj_bf  = (bf16*) alloc(BD * 2);
  float* scores    = (float*)alloc(64L * 8 * 512 * 4);
  float* maskb[4];
  for (int i = 0; i < 4; ++i) maskb[i] = (float*)alloc(BD * 4);  // mem, know, rc, wc
  float* mem_raw = (float*)alloc(BD * 4);
  bf16*  v_bf    = (bf16*) alloc(BD * 2);
  float* u_f     = (float*)alloc(64L * 1024 * 4);
  bf16*  t_bf    = (bf16*) alloc(BD * 2);
  float* g0_f    = (float*)alloc(BD * 4);
  float* g_f     = (float*)alloc(BD * 4);
  float* wc_f    = (float*)alloc(BD * 4);

  auto cast = [&](const float* s, bf16* d, long n) {
    long blocks = (n + 255) / 256; if (blocks > 4096) blocks = 4096;
    cast_kernel<<<(int)blocks, 256, 0, stream>>>(s, d, n);
  };
  const int EWB = (int)((BD + 255) / 256);

  // ---- precompute: bf16 weight copies ----
  cast(F[IN_QUESTION], q_bf, 64L * 1024);
  cast(F[P_C_POS_W], posw_bf, 12L * 512 * 1024);
  cast(F[P_C_CQ_W], cqw_bf, 512L * 1024);
  cast(F[P_R_MEM_W], memw_bf, 512L * 512);
  cast(F[P_R_KNOW_W], knoww_bf, 512L * 512);
  cast(F[P_R_CONCAT_W], rcw_bf, 512L * 1024);
  cast(F[P_R_CPM_W], rcpm_bf, 512L * 1024);
  cast(F[P_W_CONCAT_W], wcw_bf, 512L * 1024);
  cast(F[P_W_CPM_W], wcpmw_bf, 512L * 1024);
  const float* WQ[4] = {F[P_R_MM_WQ], F[P_R_KM_WQ], F[P_R_CM_WQ], F[P_W_CM_WQ]};
  const float* BQ[4] = {F[P_R_MM_BQ], F[P_R_KM_BQ], F[P_R_CM_BQ], F[P_W_CM_BQ]};
  const float* WK[4] = {F[P_R_MM_WK], F[P_R_KM_WK], F[P_R_CM_WK], F[P_W_CM_WK]};
  const float* BK[4] = {F[P_R_MM_BK], F[P_R_KM_BK], F[P_R_CM_BK], F[P_W_CM_BK]};
  for (int i = 0; i < 4; ++i) { cast(WQ[i], wq_bf[i], 512L * 512); cast(WK[i], wk_bf[i], 512L * 512); }

  // transposed copies: knowledge [b,d,K] -> know_t bf16 [b,K,d]; concat_w^T; know_w^T
  tcast_kernel<<<dim3(32, 16, 64), 256, 0, stream>>>(F[IN_KNOWLEDGE], kt, 512, 1024);
  tcast_kernel<<<dim3(32, 16, 1), 256, 0, stream>>>(F[P_R_CONCAT_W], rcwT_bf, 512, 1024);
  tcast_kernel<<<dim3(16, 16, 1), 256, 0, stream>>>(F[P_R_KNOW_W], kwT_bf, 512, 512);

  // step-invariant GEMMs (WMMA): cpm, wcpm, then the 4 MHA key matrices k = key @ wk^T + bk
  launch_gemm(stream, rcw_bf, 1024, rcpm_bf, 1024, F[P_R_CPM_B], nullptr, cpm_bf, 512, 512, 512, 1024, 1.f);
  launch_gemm(stream, wcw_bf, 1024, wcpmw_bf, 1024, F[P_W_CPM_B], nullptr, wcpm_bf, 512, 512, 512, 1024, 1.f);
  const bf16* kA[4] = {memw_bf, knoww_bf, cpm_bf, wcpm_bf};
  for (int i = 0; i < 4; ++i)
    launch_gemm(stream, kA[i], 512, wk_bf[i], 512, BK[i], nullptr, kX_bf[i], 512, 512, 512, 512, 1.f);

  init_state_kernel<<<EWB, 256, 0, stream>>>(F[P_CONTROL0], F[P_MEM0], control, memory_f, cqA, wcA);

  // ---- 12 MAC steps ----
  for (int i = 0; i < 12; ++i) {
    // ControlUnit: pa = question @ pos_w[i]^T + pos_b[i]  -> cqA[:,512:]
    launch_gemm(stream, q_bf, 1024, posw_bf + (long)i * 512 * 1024, 1024,
                F[P_C_POS_B] + (long)i * 512, nullptr, cqA + 512, 1024, 64, 512, 1024, 1.f);
    // cq = [control_prev | pa] @ cq_w^T + cq_b
    launch_gemm(stream, cqA, 1024, cqw_bf, 1024, F[P_C_CQ_B], cq_f, nullptr, 512, 64, 512, 1024, 1.f);
    // control = softmax((cq*ctx)@attn_w) . ctx   (also writes control_bf and cqA[:, :512])
    control_attn_kernel<<<64, 256, 0, stream>>>(cq_f, F[IN_CONTEXT], F[P_C_ATTN_W], F[P_C_ATTN_B],
                                                control, control_bf, cqA);
    // 4 MHA masks of control: q = 0.125*(control@wq^T + bq); scores per head; softmax+mean
    for (int mm = 0; mm < 4; ++mm) {
      launch_gemm(stream, control_bf, 512, wq_bf[mm], 512, BQ[mm], nullptr, qproj_bf, 512,
                  64, 512, 512, 0.125f);
      launch_gemm(stream, qproj_bf, 512, kX_bf[mm], 512, nullptr, scores, nullptr, 4096,
                  64, 512, 64, 1.f, /*Z=*/8, /*saz=*/64, /*sbz=*/64, /*scz=*/512);
      mask_softmax_kernel<<<64, 256, 0, stream>>>(scores, maskb[mm]);
    }
    // ReadUnit fold: mem_raw = memory_prev @ mem_w^T + mem_b
    launch_gemm(stream, wcA + 512, 1024, memw_bf, 512, F[P_R_MEM_B], mem_raw, nullptr, 512,
                64, 512, 512, 1.f);
    mk_v_kernel<<<EWB, 256, 0, stream>>>(control, maskb[2], F[P_R_ATTN_W], v_bf);
    launch_gemm(stream, v_bf, 512, rcwT_bf, 512, nullptr, u_f, nullptr, 1024, 64, 1024, 512, 1.f);
    mk_t_kernel<<<EWB, 256, 0, stream>>>(maskb[1], mem_raw, maskb[0], u_f, t_bf);
    launch_gemm(stream, t_bf, 512, kwT_bf, 512, nullptr, g0_f, nullptr, 512, 64, 512, 512, 1.f);
    add_g_kernel<<<EWB, 256, 0, stream>>>(g0_f, u_f, g_f);
    // streaming: logits = know_t.g, softmax over K=1024, read -> wcA[:, :512]
    read_unit_kernel<<<64, 256, 0, stream>>>(kt, g_f, wcA);
    // WriteUnit: wc = [read | memory_prev] @ w.concat_w^T + b ; memory = wc * wc_mask
    launch_gemm(stream, wcA, 1024, wcw_bf, 1024, F[P_W_CONCAT_B], wc_f, nullptr, 512,
                64, 512, 1024, 1.f);
    mem_update_kernel<<<EWB, 256, 0, stream>>>(wc_f, maskb[3], memory_f, wcA);
  }

  hipMemcpyAsync(d_out, memory_f, BD * sizeof(float), hipMemcpyDeviceToDevice, stream);
}
